// GAT_GCN_77498389889614
// MI455X (gfx1250) — compile-verified, matrix operand order checked by reading, not run on hardware
//
#include <hip/hip_runtime.h>
#include <hip/hip_bf16.h>
#include <math.h>

#define N_NODES 40000
#define N_EDGES 640000
#define EE_EDGES (N_EDGES + N_NODES)

typedef __attribute__((ext_vector_type(16))) __bf16 v16bf;
typedef __attribute__((ext_vector_type(8)))  __bf16 v8bf;
typedef __attribute__((ext_vector_type(8)))  float  v8f;

// native f32->bf16 convert (v_cvt_*_bf16_f32, RTNE) instead of integer emulation
__device__ __forceinline__ __bf16 to_bf16(float f) { return (__bf16)f; }

// ---------------------------------------------------------------- utilities
__global__ void fill_f32(float* __restrict__ p, float v, int n) {
  int i = blockIdx.x * blockDim.x + threadIdx.x;
  if (i < n) p[i] = v;
}

// W[K,NCOL] f32 -> Bt[NCOL,K] bf16 (transposed, so B fragments are contiguous)
__global__ void convert_wt(const float* __restrict__ W, __bf16* __restrict__ Bt,
                           int K, int NCOL) {
  int i = blockIdx.x * blockDim.x + threadIdx.x;
  if (i >= K * NCOL) return;
  int n = i / K, k = i - n * K;
  Bt[i] = to_bf16(W[k * NCOL + n]);
}

// ---------------------------------------------------------------- WMMA GEMM
// C[M,NCOL] = A[M,K](f32) x Bt[NCOL,K](bf16), optional bias.
// block = 128 threads (4 waves); wave w -> rows blockIdx.x*64 + 16w;
// blockIdx.y -> 64-column slab (4 WMMA tiles per wave, A fragment reused).
// Requires M%64==0, K%32==0, NCOL%64==0 -> no predication, EXEC all ones.
__global__ void __launch_bounds__(128)
gemm_bf16_wmma(const float* __restrict__ A, const __bf16* __restrict__ Bt,
               float* __restrict__ C, const float* __restrict__ bias,
               int K, int NCOL) {
  const int wave = threadIdx.x >> 5;
  const int lane = threadIdx.x & 31;
  const int half = lane >> 4;
  const int l15  = lane & 15;
  const int row0 = blockIdx.x * 64 + wave * 16;
  const int col0 = blockIdx.y * 64;

  v8f acc0 = {}, acc1 = {}, acc2 = {}, acc3 = {};
  const float* Arow = A + (size_t)(row0 + l15) * K;

  for (int kk = 0; kk < K; kk += 32) {
    const int kb0 = kk + half * 8;      // A 16x32 bf16 fragment layout (ISA 7.12.2)
    const int kb1 = kb0 + 16;
    v16bf a;
    {
      const float4 f0 = *(const float4*)(Arow + kb0);
      const float4 f1 = *(const float4*)(Arow + kb0 + 4);
      const float4 f2 = *(const float4*)(Arow + kb1);
      const float4 f3 = *(const float4*)(Arow + kb1 + 4);
      a[0]=to_bf16(f0.x);  a[1]=to_bf16(f0.y);  a[2]=to_bf16(f0.z);  a[3]=to_bf16(f0.w);
      a[4]=to_bf16(f1.x);  a[5]=to_bf16(f1.y);  a[6]=to_bf16(f1.z);  a[7]=to_bf16(f1.w);
      a[8]=to_bf16(f2.x);  a[9]=to_bf16(f2.y);  a[10]=to_bf16(f2.z); a[11]=to_bf16(f2.w);
      a[12]=to_bf16(f3.x); a[13]=to_bf16(f3.y); a[14]=to_bf16(f3.z); a[15]=to_bf16(f3.w);
    }
#define WMMA_TILE(T, ACC) { \
    const __bf16* Bcol = Bt + (size_t)(col0 + (T)*16 + l15) * K;               \
    const v8bf b0 = *(const v8bf*)(Bcol + kb0);                                \
    const v8bf b1 = *(const v8bf*)(Bcol + kb1);                                \
    v16bf b;                                                                   \
    b[0]=b0[0]; b[1]=b0[1]; b[2]=b0[2]; b[3]=b0[3];                            \
    b[4]=b0[4]; b[5]=b0[5]; b[6]=b0[6]; b[7]=b0[7];                            \
    b[8]=b1[0]; b[9]=b1[1]; b[10]=b1[2]; b[11]=b1[3];                          \
    b[12]=b1[4]; b[13]=b1[5]; b[14]=b1[6]; b[15]=b1[7];                        \
    ACC = __builtin_amdgcn_wmma_f32_16x16x32_bf16(false, a, false, b,          \
                                                  (short)0, ACC, false, false); }
    WMMA_TILE(0, acc0)
    WMMA_TILE(1, acc1)
    WMMA_TILE(2, acc2)
    WMMA_TILE(3, acc3)
#undef WMMA_TILE
  }

#define STORE_TILE(T, ACC) { \
    const int col = col0 + (T)*16 + l15;                                       \
    const float bv = bias ? bias[col] : 0.0f;                                  \
    float* Cp = C + (size_t)(row0 + half * 8) * NCOL + col;                    \
    _Pragma("unroll")                                                          \
    for (int j = 0; j < 8; ++j) Cp[(size_t)j * NCOL] = ACC[j] + bv; }
  STORE_TILE(0, acc0)
  STORE_TILE(1, acc1)
  STORE_TILE(2, acc2)
  STORE_TILE(3, acc3)
#undef STORE_TILE
}

// ---------------------------------------------------------------- GAT pieces
// one wave per (node, head): dot(xh[n,h,:], att_src[h,:]) and att_dst
__global__ void att_dot(const float* __restrict__ xh,
                        const float* __restrict__ att_src,
                        const float* __restrict__ att_dst,
                        float* __restrict__ as_, float* __restrict__ ad_,
                        int H, int Cc) {
  int wid  = (blockIdx.x * blockDim.x + threadIdx.x) >> 5;
  int lane = threadIdx.x & 31;
  if (wid >= N_NODES * H) return;
  int h = wid % H;
  const float* xp = xh + (size_t)wid * Cc;   // xh flat [N*H, C]
  float ss = 0.f, sd = 0.f;
  for (int c = lane; c < Cc; c += 32) {
    float v = xp[c];
    ss += v * att_src[h * Cc + c];
    sd += v * att_dst[h * Cc + c];
  }
  for (int o = 16; o > 0; o >>= 1) {
    ss += __shfl_xor(ss, o, 32);
    sd += __shfl_xor(sd, o, 32);
  }
  if (lane == 0) { as_[wid] = ss; ad_[wid] = sd; }
}

__device__ __forceinline__ void atomicMaxF(float* addr, float v) {
  int iv = __float_as_int(v);
  if (iv >= 0) atomicMax((int*)addr, iv);
  else         atomicMin((unsigned int*)addr, (unsigned int)iv);
}

__global__ void edge_max(const int* __restrict__ src, const int* __restrict__ dst,
                         const float* __restrict__ as_, const float* __restrict__ ad_,
                         float* __restrict__ mbuf, int H) {
  int i = blockIdx.x * blockDim.x + threadIdx.x;
  if (i >= EE_EDGES * H) return;
  int e = i / H, h = i - e * H;
  int s = (e < N_EDGES) ? src[e] : (e - N_EDGES);
  int d = (e < N_EDGES) ? dst[e] : (e - N_EDGES);
  float x  = as_[s * H + h] + ad_[d * H + h];
  float lr = x > 0.f ? x : 0.2f * x;
  atomicMaxF(&mbuf[d * H + h], lr);
}

__global__ void edge_exp(const int* __restrict__ src, const int* __restrict__ dst,
                         const float* __restrict__ as_, const float* __restrict__ ad_,
                         const float* __restrict__ mbuf, float* __restrict__ abuf,
                         float* __restrict__ sbuf, int H) {
  int i = blockIdx.x * blockDim.x + threadIdx.x;
  if (i >= EE_EDGES * H) return;
  int e = i / H, h = i - e * H;
  int s = (e < N_EDGES) ? src[e] : (e - N_EDGES);
  int d = (e < N_EDGES) ? dst[e] : (e - N_EDGES);
  float x  = as_[s * H + h] + ad_[d * H + h];
  float lr = x > 0.f ? x : 0.2f * x;
  float m  = mbuf[d * H + h];
  if (!(m > -1e37f)) m = 0.f;       // where(isfinite(m), m, 0)
  float a = __expf(lr - m);
  abuf[i] = a;
  atomicAdd(&sbuf[d * H + h], a);
}

// one wave per edge; lanes cover H*C features
__global__ void edge_aggr(const int* __restrict__ src, const int* __restrict__ dst,
                          const float* __restrict__ abuf, const float* __restrict__ sbuf,
                          const float* __restrict__ xh, float* __restrict__ outp,
                          int H, int Cc) {
  int wid  = (blockIdx.x * blockDim.x + threadIdx.x) >> 5;
  int lane = threadIdx.x & 31;
  if (wid >= EE_EDGES) return;
  int s = (wid < N_EDGES) ? src[wid] : (wid - N_EDGES);
  int d = (wid < N_EDGES) ? dst[wid] : (wid - N_EDGES);
  float w0 = abuf[(size_t)wid * H]     / (sbuf[d * H]     + 1e-16f);
  float w1 = (H > 1) ? abuf[(size_t)wid * H + 1] / (sbuf[d * H + 1] + 1e-16f) : 0.f;
  int HC = H * Cc;
  const float* xp = xh   + (size_t)s * HC;
  float*       op = outp + (size_t)d * HC;
  for (int c = lane; c < HC; c += 32) {
    float w = (c < Cc) ? w0 : w1;     // H <= 2
    atomicAdd(&op[c], w * xp[c]);
  }
}

// mode: 0 = +bias, 1 = relu(+bias), 2 = elu(+bias)
__global__ void bias_act(float* __restrict__ p, const float* __restrict__ bias,
                         int total, int Cc, int mode) {
  int i = blockIdx.x * blockDim.x + threadIdx.x;
  if (i >= total) return;
  float v = p[i] + bias[i % Cc];
  if (mode == 1)      v = v > 0.f ? v : 0.f;
  else if (mode == 2) v = v > 0.f ? v : (__expf(v) - 1.f);
  p[i] = v;
}

// ---------------------------------------------------------------- GCN pieces
__global__ void deg_acc(const int* __restrict__ dst, float* __restrict__ deg) {
  int i = blockIdx.x * blockDim.x + threadIdx.x;
  if (i >= EE_EDGES) return;
  int d = (i < N_EDGES) ? dst[i] : (i - N_EDGES);
  atomicAdd(&deg[d], 1.0f);
}

__global__ void dinv_k(float* __restrict__ deg) {
  int i = blockIdx.x * blockDim.x + threadIdx.x;
  if (i >= N_NODES) return;
  deg[i] = rsqrtf(fmaxf(deg[i], 1.0f));
}

__global__ void gcn_scatter(const int* __restrict__ src, const int* __restrict__ dst,
                            const float* __restrict__ dinv, const float* __restrict__ xw,
                            float* __restrict__ outp, int Cc) {
  int wid  = (blockIdx.x * blockDim.x + threadIdx.x) >> 5;
  int lane = threadIdx.x & 31;
  if (wid >= EE_EDGES) return;
  int s = (wid < N_EDGES) ? src[wid] : (wid - N_EDGES);
  int d = (wid < N_EDGES) ? dst[wid] : (wid - N_EDGES);
  float w = dinv[s] * dinv[d];
  const float* xp = xw   + (size_t)s * Cc;
  float*       op = outp + (size_t)d * Cc;
  for (int c = lane; c < Cc; c += 32) atomicAdd(&op[c], w * xp[c]);
}

// xcat[n, 0:C] = wc*gcn[n], xcat[n, C:2C] = wt*gat[n]
__global__ void concat_k(const float* __restrict__ gcn, const float* __restrict__ gat,
                         const float* __restrict__ wc, const float* __restrict__ wt,
                         float* __restrict__ xcat, int Cc) {
  int i = blockIdx.x * blockDim.x + threadIdx.x;
  int total = N_NODES * 2 * Cc;
  if (i >= total) return;
  int n = i / (2 * Cc);
  int c = i - n * (2 * Cc);
  xcat[i] = (c < Cc) ? (wc[0] * gcn[(size_t)n * Cc + c])
                     : (wt[0] * gat[(size_t)n * Cc + (c - Cc)]);
}

// ---------------------------------------------------------------- launcher
extern "C" void kernel_launch(void* const* d_in, const int* in_sizes, int n_in,
                              void* d_out, int out_size, void* d_ws, size_t ws_size,
                              hipStream_t stream) {
  (void)in_sizes; (void)n_in; (void)out_size; (void)ws_size;
  const float* x      = (const float*)d_in[0];
  const int*   ei     = (const int*)d_in[1];
  const float* gat1_W  = (const float*)d_in[2];
  const float* gat1_as = (const float*)d_in[3];
  const float* gat1_ad = (const float*)d_in[4];
  const float* gat1_b  = (const float*)d_in[5];
  const float* gat2_W  = (const float*)d_in[6];
  const float* gat2_as = (const float*)d_in[7];
  const float* gat2_ad = (const float*)d_in[8];
  const float* gat2_b  = (const float*)d_in[9];
  const float* gcn1_W  = (const float*)d_in[10];
  const float* gcn1_b  = (const float*)d_in[11];
  const float* gcn2_W  = (const float*)d_in[12];
  const float* gcn2_b  = (const float*)d_in[13];
  const float* lin1_W  = (const float*)d_in[14];
  const float* lin1_b  = (const float*)d_in[15];
  const float* lin2_W  = (const float*)d_in[16];
  const float* lin2_b  = (const float*)d_in[17];
  const float* lin3_W  = (const float*)d_in[18];
  const float* lin3_b  = (const float*)d_in[19];
  const float* wT      = (const float*)d_in[20];
  const float* wC      = (const float*)d_in[21];
  float* out = (float*)d_out;

  const int* src = ei;
  const int* dst = ei + N_EDGES;

  float*  ws  = (float*)d_ws;
  __bf16* wbt = (__bf16*)d_ws;

  // bf16 weight offsets (elements)
  const size_t BT_GAT1 = 0, BT_GAT2 = 32768, BT_GCN1 = 49152, BT_GCN2 = 65536,
               BT_LIN1 = 73728, BT_LIN2 = 106496, BT_LIN3 = 139264; // end 147456 bf16
  // float region offsets (bf16 area = 294912 B < 160000 floats)
  const size_t OFF_R1  = 160000;
  const size_t OFF_R2  = OFF_R1  + (size_t)N_NODES * 256;
  const size_t OFF_R3  = OFF_R2  + (size_t)N_NODES * 256;
  const size_t OFF_AB  = OFF_R3  + (size_t)N_NODES * 64;
  const size_t OFF_AS1 = OFF_AB  + (size_t)EE_EDGES * 2;
  const size_t OFF_AD1 = OFF_AS1 + (size_t)N_NODES * 2;
  const size_t OFF_M1  = OFF_AD1 + (size_t)N_NODES * 2;
  const size_t OFF_S1  = OFF_M1  + (size_t)N_NODES * 2;
  const size_t OFF_AS2 = OFF_S1  + (size_t)N_NODES * 2;
  const size_t OFF_AD2 = OFF_AS2 + (size_t)N_NODES;
  const size_t OFF_M2  = OFF_AD2 + (size_t)N_NODES;
  const size_t OFF_S2  = OFF_M2  + (size_t)N_NODES;
  const size_t OFF_DEG = OFF_S2  + (size_t)N_NODES;

  float* R1 = ws + OFF_R1;                    // N*256
  float* R2 = ws + OFF_R2;                    // N*256
  float* R3 = ws + OFF_R3;                    // N*64
  float* AB = ws + OFF_AB;
  float* AS1 = ws + OFF_AS1; float* AD1 = ws + OFF_AD1;
  float* M1  = ws + OFF_M1;  float* S1  = ws + OFF_S1;
  float* AS2 = ws + OFF_AS2; float* AD2 = ws + OFF_AD2;
  float* M2  = ws + OFF_M2;  float* S2  = ws + OFF_S2;
  float* DEG = ws + OFF_DEG;

  // buffer timeline (regions reused, strictly in-order on `stream`)
  float* xh1     = R1;
  float* agg1    = R2;                                   // -> x_gat (elu, in place)
  float* xh2     = R1;
  float* agg2    = R1 + (size_t)N_NODES * 64;            // -> gat2_out
  float* xw1     = R2;
  float* agg_g1  = R2 + (size_t)N_NODES * 128;           // -> x_gcn1 (relu)
  float* xw2     = R1;
  float* agg_g2  = R3;                                   // -> gcn2_out
  float* xcat    = R2;
  float* h1      = R1;
  float* h2      = R2 + (size_t)N_NODES * 128;

  const float NEG_INF = -__builtin_huge_valf();
  auto cdiv = [](long long a, long long b) { return (int)((a + b - 1) / b); };

  // ---- weight conversion (f32 -> transposed bf16) ----
  convert_wt<<<cdiv(128*256,256),256,0,stream>>>(gat1_W, wbt+BT_GAT1, 128, 256);
  convert_wt<<<cdiv(256*64 ,256),256,0,stream>>>(gat2_W, wbt+BT_GAT2, 256, 64);
  convert_wt<<<cdiv(128*128,256),256,0,stream>>>(gcn1_W, wbt+BT_GCN1, 128, 128);
  convert_wt<<<cdiv(128*64 ,256),256,0,stream>>>(gcn2_W, wbt+BT_GCN2, 128, 64);
  convert_wt<<<cdiv(128*256,256),256,0,stream>>>(lin1_W, wbt+BT_LIN1, 128, 256);
  convert_wt<<<cdiv(256*128,256),256,0,stream>>>(lin2_W, wbt+BT_LIN2, 256, 128);
  convert_wt<<<cdiv(128*64 ,256),256,0,stream>>>(lin3_W, wbt+BT_LIN3, 128, 64);

  // ---- GAT layer 1 (H=2, C=128) ----
  gemm_bf16_wmma<<<dim3(625,4),128,0,stream>>>(x, wbt+BT_GAT1, xh1, nullptr, 128, 256);
  att_dot<<<cdiv((long long)N_NODES*2*32,256),256,0,stream>>>(xh1, gat1_as, gat1_ad, AS1, AD1, 2, 128);
  fill_f32<<<cdiv(N_NODES*2,256),256,0,stream>>>(M1, NEG_INF, N_NODES*2);
  fill_f32<<<cdiv(N_NODES*2,256),256,0,stream>>>(S1, 0.f, N_NODES*2);
  fill_f32<<<cdiv((long long)N_NODES*256,256),256,0,stream>>>(agg1, 0.f, N_NODES*256);
  edge_max<<<cdiv((long long)EE_EDGES*2,256),256,0,stream>>>(src, dst, AS1, AD1, M1, 2);
  edge_exp<<<cdiv((long long)EE_EDGES*2,256),256,0,stream>>>(src, dst, AS1, AD1, M1, AB, S1, 2);
  edge_aggr<<<cdiv((long long)EE_EDGES*32,256),256,0,stream>>>(src, dst, AB, S1, xh1, agg1, 2, 128);
  bias_act<<<cdiv((long long)N_NODES*256,256),256,0,stream>>>(agg1, gat1_b, N_NODES*256, 256, 2); // ELU

  // ---- GAT layer 2 (H=1, C=64) ----
  gemm_bf16_wmma<<<dim3(625,1),128,0,stream>>>(agg1, wbt+BT_GAT2, xh2, nullptr, 256, 64);
  att_dot<<<cdiv((long long)N_NODES*32,256),256,0,stream>>>(xh2, gat2_as, gat2_ad, AS2, AD2, 1, 64);
  fill_f32<<<cdiv(N_NODES,256),256,0,stream>>>(M2, NEG_INF, N_NODES);
  fill_f32<<<cdiv(N_NODES,256),256,0,stream>>>(S2, 0.f, N_NODES);
  fill_f32<<<cdiv((long long)N_NODES*64,256),256,0,stream>>>(agg2, 0.f, N_NODES*64);
  edge_max<<<cdiv(EE_EDGES,256),256,0,stream>>>(src, dst, AS2, AD2, M2, 1);
  edge_exp<<<cdiv(EE_EDGES,256),256,0,stream>>>(src, dst, AS2, AD2, M2, AB, S2, 1);
  edge_aggr<<<cdiv((long long)EE_EDGES*32,256),256,0,stream>>>(src, dst, AB, S2, xh2, agg2, 1, 64);
  bias_act<<<cdiv((long long)N_NODES*64,256),256,0,stream>>>(agg2, gat2_b, N_NODES*64, 64, 0);

  // ---- GCN layer 1 (C=128, ReLU) ----
  gemm_bf16_wmma<<<dim3(625,2),128,0,stream>>>(x, wbt+BT_GCN1, xw1, nullptr, 128, 128);
  fill_f32<<<cdiv(N_NODES,256),256,0,stream>>>(DEG, 0.f, N_NODES);
  deg_acc<<<cdiv(EE_EDGES,256),256,0,stream>>>(dst, DEG);
  dinv_k<<<cdiv(N_NODES,256),256,0,stream>>>(DEG);
  fill_f32<<<cdiv((long long)N_NODES*128,256),256,0,stream>>>(agg_g1, 0.f, N_NODES*128);
  gcn_scatter<<<cdiv((long long)EE_EDGES*32,256),256,0,stream>>>(src, dst, DEG, xw1, agg_g1, 128);
  bias_act<<<cdiv((long long)N_NODES*128,256),256,0,stream>>>(agg_g1, gcn1_b, N_NODES*128, 128, 1);

  // ---- GCN layer 2 (C=64) ----
  gemm_bf16_wmma<<<dim3(625,1),128,0,stream>>>(agg_g1, wbt+BT_GCN2, xw2, nullptr, 128, 64);
  fill_f32<<<cdiv((long long)N_NODES*64,256),256,0,stream>>>(agg_g2, 0.f, N_NODES*64);
  gcn_scatter<<<cdiv((long long)EE_EDGES*32,256),256,0,stream>>>(src, dst, DEG, xw2, agg_g2, 64);
  bias_act<<<cdiv((long long)N_NODES*64,256),256,0,stream>>>(agg_g2, gcn2_b, N_NODES*64, 64, 0);

  // ---- concat + MLP head ----
  concat_k<<<cdiv((long long)N_NODES*128,256),256,0,stream>>>(agg_g2, agg2, wC, wT, xcat, 64);
  gemm_bf16_wmma<<<dim3(625,4),128,0,stream>>>(xcat, wbt+BT_LIN1, h1,  lin1_b, 128, 256);
  gemm_bf16_wmma<<<dim3(625,2),128,0,stream>>>(h1,   wbt+BT_LIN2, h2,  lin2_b, 256, 128);
  gemm_bf16_wmma<<<dim3(625,1),128,0,stream>>>(h2,   wbt+BT_LIN3, out, lin3_b, 128, 64);
}
